// QuantizedConv2d_53730040873276
// MI455X (gfx1250) — 8-bit matrix operand order repaired and checked by reading, compiled, not run on hardware
//
#include <hip/hip_runtime.h>
#include <stdint.h>

typedef int   v8i __attribute__((ext_vector_type(8)));
typedef int   v4i __attribute__((ext_vector_type(4)));
typedef int   v2i __attribute__((ext_vector_type(2)));
typedef float v4f __attribute__((ext_vector_type(4)));
typedef unsigned int v4u __attribute__((ext_vector_type(4)));

#define CIN   128
#define COUT  128
#define WDIM  56
#define HW    3136
#define NPIX  200704

#define BROW  144
#define BLDS  (9 * 64 * BROW)

#if defined(__has_builtin)
#  if __has_builtin(__builtin_amdgcn_tensor_load_to_lds)
#    define HAVE_TDM 1
#  endif
#endif
#ifndef HAVE_TDM
#  define HAVE_TDM 0
#endif

__global__ __launch_bounds__(256)
void repack_weight_kernel(const int* __restrict__ w, int8_t* __restrict__ wT) {
    int idx = blockIdx.x * 256 + threadIdx.x;
    if (idx >= 9 * COUT * CIN) return;
    int tap = idx / (COUT * CIN);
    int r   = idx % (COUT * CIN);
    int co  = r / CIN;
    int ci  = r % CIN;
    wT[idx] = (int8_t)w[(co * CIN + ci) * 9 + tap];
}

__global__ __launch_bounds__(256)
void repack_input_kernel(const int* __restrict__ x, int8_t* __restrict__ xT) {
    __shared__ int8_t tile[32][33];
    const int p0 = blockIdx.x * 32;
    const int c0 = blockIdx.y * 32;
    const int n  = p0 / HW;
    const int s0 = p0 % HW;
    const int t  = threadIdx.x;
    #pragma unroll
    for (int i = 0; i < 4; ++i) {
        int lin = t + i * 256;
        int ci = lin >> 5, si = lin & 31;
        tile[ci][si] = (int8_t)x[(size_t)(n * CIN + c0 + ci) * HW + s0 + si];
    }
    __syncthreads();
    #pragma unroll
    for (int i = 0; i < 4; ++i) {
        int lin = t + i * 256;
        int si = lin >> 5, ci = lin & 31;
        xT[(size_t)(p0 + si) * CIN + c0 + ci] = tile[ci][si];
    }
}

__global__ __launch_bounds__(256)
void qconv_wmma_kernel(const int8_t* __restrict__ xT,
                       const int8_t* __restrict__ wT,
                       const int*    __restrict__ bias,
                       const float*  __restrict__ in_scale,
                       const float*  __restrict__ w_scale,
                       const float*  __restrict__ out_scale,
                       const int*    __restrict__ in_zp,
                       const int*    __restrict__ out_zp,
                       float*        __restrict__ out) {
    __shared__ int8_t wLds[BLDS];

    const int lane = threadIdx.x & 31;
    const int wave = threadIdx.x >> 5;
    const int pw   = wave & 3;
    const int cw   = wave >> 2;
    const int lidx = lane & 15;
    const int hi   = lane >> 4;

    const int pixWave = blockIdx.x * 64 + pw * 16;
    const int n       = pixWave / HW;
    const int s0      = pixWave % HW;
    const int coutBlk = blockIdx.y * 64;

#if HAVE_TDM
    {
        const uint32_t ldsAddr = (uint32_t)(uintptr_t)(&wLds[0]);
        const uint64_t ga = (uint64_t)(uintptr_t)(wT + (size_t)coutBlk * CIN);
        v4u g0 = { 1u,
                   ldsAddr,
                   (uint32_t)(ga & 0xFFFFFFFFu),
                   (uint32_t)((ga >> 32) & 0x01FFFFFFu) | (2u << 30) };
        v8i g1 = { (int)((1u << 20) | (4u << 22) | (3u << 25)),
                   (int)(128u << 16),
                   (int)(128u << 16),
                   (int)(128u << 16),
                   (int)(64u | (9u << 16)),
                   (int)128,
                   (int)(0x4000u << 16),
                   0 };
        v4i g2 = { 9, 0, 0, 0 };
        v4i g3 = { 0, 0, 0, 0 };
        v8i g4 = { 0, 0, 0, 0, 0, 0, 0, 0 };
        if (wave == 0) {
            __builtin_amdgcn_tensor_load_to_lds(g0, g1, g2, g3, g4, 0);
            __builtin_amdgcn_s_wait_tensorcnt(0);
        }
    }
#else
    for (int c = threadIdx.x; c < 9 * 64 * 8; c += 256) {
        int row = c >> 3, off = (c & 7) * 16;
        int tap = row >> 6, co = row & 63;
        *(v4i*)(wLds + row * BROW + off) =
            *(const v4i*)(wT + (size_t)tap * COUT * CIN +
                          (size_t)(coutBlk + co) * CIN + off);
    }
#endif
    __syncthreads();

    const int sPix = s0 + lidx;
    const int py   = sPix / WDIM;
    const int px   = sPix % WDIM;

    const int      izp = in_zp[0];
    const uint32_t zpb = (uint32_t)(izp & 0xFF) * 0x01010101u;

    const int rowA = cw * 32 + lidx;

    v8i acc0 = {};
    v8i acc1 = {};

    #pragma unroll 3
    for (int tap = 0; tap < 9; ++tap) {
        const int ky = tap / 3, kx = tap % 3;
        const int yy = py + ky - 1;
        const int xx = px + kx - 1;
        const bool valid = (yy >= 0) && (yy < WDIM) && (xx >= 0) && (xx < WDIM);
        const int yc = yy < 0 ? 0 : (yy > WDIM - 1 ? WDIM - 1 : yy);
        const int xc = xx < 0 ? 0 : (xx > WDIM - 1 ? WDIM - 1 : xx);
        const int8_t* aBase = xT + (size_t)(n * HW + yc * WDIM + xc) * CIN;
        const int8_t* bTap  = wLds + tap * (64 * BROW);

        #pragma unroll
        for (int kc = 0; kc < CIN; kc += 64) {
            union { v2i d[4]; v8i v; } A;
            const int kA = kc + hi * 16;
            #pragma unroll
            for (int j = 0; j < 4; ++j) {
                v2i d = *(const v2i*)(aBase + kA + (j & 1) * 8 + (j >> 1) * 32);
                if (!valid) { d.x = (int)zpb; d.y = (int)zpb; }
                A.d[j] = d;
            }

            const int kB = kc + hi * 16;
            union { v4i q[2]; v8i v; } B0, B1;
            #pragma unroll
            for (int c = 0; c < 2; ++c) {
                B0.q[c] = *(const v4i*)(bTap + rowA * BROW + kB + c * 32);
                B1.q[c] = *(const v4i*)(bTap + (rowA + 16) * BROW + kB + c * 32);
            }

            acc0 = __builtin_amdgcn_wmma_i32_16x16x64_iu8(
                       true, A.v, true, B0.v, acc0, false, false);
            acc1 = __builtin_amdgcn_wmma_i32_16x16x64_iu8(
                       true, A.v, true, B1.v, acc1, false, false);
        }
    }

    const float is  = in_scale[0];
    const float os  = out_scale[0];
    const float ozp = (float)out_zp[0];
    const int co0 = coutBlk + cw * 32 + lidx;
    const int co1 = co0 + 16;
    const float sc0 = is * w_scale[co0] / os;
    const float sc1 = is * w_scale[co1] / os;
    const float b0  = (float)bias[co0];
    const float b1  = (float)bias[co1];

    const int sRow = s0 + hi * 8;
    float* o0 = out + ((size_t)(n * COUT + co0)) * HW + sRow;
    float* o1 = out + ((size_t)(n * COUT + co1)) * HW + sRow;

    float r0[8], r1[8];
    #pragma unroll
    for (int v = 0; v < 8; ++v) {
        float f0 = ((float)acc0[v] + b0) * sc0 + ozp;
        float f1 = ((float)acc1[v] + b1) * sc1 + ozp;
        f0 = rintf(f0); f1 = rintf(f1);
        r0[v] = fminf(fmaxf(f0, -128.0f), 127.0f);
        r1[v] = fminf(fmaxf(f1, -128.0f), 127.0f);
    }
    *(v4f*)(o0)     = (v4f){r0[0], r0[1], r0[2], r0[3]};
    *(v4f*)(o0 + 4) = (v4f){r0[4], r0[5], r0[6], r0[7]};
    *(v4f*)(o1)     = (v4f){r1[0], r1[1], r1[2], r1[3]};
    *(v4f*)(o1 + 4) = (v4f){r1[4], r1[5], r1[6], r1[7]};
}

extern "C" void kernel_launch(void* const* d_in, const int* in_sizes, int n_in,
                              void* d_out, int out_size, void* d_ws, size_t ws_size,
                              hipStream_t stream) {
    const int*   x    = (const int*)d_in[0];
    const int*   w    = (const int*)d_in[1];
    const int*   bias = (const int*)d_in[2];
    const float* is   = (const float*)d_in[3];
    const float* wsc  = (const float*)d_in[4];
    const float* os   = (const float*)d_in[5];
    const int*   izp  = (const int*)d_in[6];
    const int*   ozp  = (const int*)d_in[7];

    int8_t* xT = (int8_t*)d_ws;
    int8_t* wT = xT + (size_t)NPIX * CIN;

    repack_weight_kernel<<<(9 * COUT * CIN + 255) / 256, 256, 0, stream>>>(w, wT);

    dim3 gi(NPIX / 32, CIN / 32);
    repack_input_kernel<<<gi, 256, 0, stream>>>(x, xT);

    dim3 gm(NPIX / 64, COUT / 64);
    qconv_wmma_kernel<<<gm, 256, 0, stream>>>(xT, wT, bias, is, wsc, os,
                                              izp, ozp, (float*)d_out);
}
